// HybridGATLSTM_for_predict_8693013807247
// MI455X (gfx1250) — matrix operand order repaired and checked by reading, not compiled, not run on hardware
//
#include <hip/hip_runtime.h>
#include <math.h>

// ---------------------------------------------------------------------------
// Problem constants
// ---------------------------------------------------------------------------
#define NNODES 500
#define GATH   32
#define HIDDEN 128
#define BB     16
#define SS     64
#define EE     8000
#define ET     (EE + NNODES)          // edges + self loops = 8500
#define BS     (BB * SS)              // 1024
#define KDIM   (NNODES * GATH)        // 16000
#define GDIM   (4 * HIDDEN)           // 512
#define KSPLIT 5
#define KH     (KDIM / KSPLIT)        // 3200 -> 100 K-steps of 32
#define HSTR   136                    // padded bf16 row stride for h in LDS
#define TPAD   40                     // padded LDS row stride (bf16) for GEMM tiles

typedef __attribute__((ext_vector_type(16))) __bf16 v16bf;
typedef __attribute__((ext_vector_type(8)))  float  v8f;

union ABFrag { uint4 u[2]; v16bf v; };

// float -> bf16 (round to nearest even)
__device__ __forceinline__ unsigned short f2bf(float f) {
    unsigned int u = __float_as_uint(f);
    u += 0x7FFFu + ((u >> 16) & 1u);
    return (unsigned short)(u >> 16);
}

__device__ __forceinline__ float sigmoidf_(float x) {
    return 1.0f / (1.0f + expf(-x));
}

// atomic float max via signed/unsigned int trick (valid with -inf init)
__device__ __forceinline__ void atomicMaxF(float* addr, float val) {
    if (val >= 0.0f) atomicMax((int*)addr, __float_as_int(val));
    else             atomicMin((unsigned int*)addr, (unsigned int)__float_as_int(val));
}

// ---------------------------------------------------------------------------
// K0: init m/den/num, transpose x -> xt[n*BS+bs], compute ws/wd scalars
// ---------------------------------------------------------------------------
__global__ __launch_bounds__(256) void init_kernel(
    const float* __restrict__ x, const float* __restrict__ gat_w,
    const float* __restrict__ a_src, const float* __restrict__ a_dst,
    float* __restrict__ xt, float* __restrict__ m,
    float* __restrict__ den, float* __restrict__ num,
    float* __restrict__ scal)
{
    int idx = blockIdx.x * blockDim.x + threadIdx.x;
    const int total = NNODES * BS;
    if (idx < total) {
        int bs = idx & (BS - 1);
        int n  = idx >> 10;
        xt[idx]  = x[bs * NNODES + n];
        m[idx]   = -__builtin_inff();
        den[idx] = 0.0f;
        num[idx] = 0.0f;
    }
    if (idx == 0) {
        float ws = 0.0f, wd = 0.0f;
        for (int f = 0; f < GATH; ++f) {
            ws += gat_w[f] * a_src[f];
            wd += gat_w[f] * a_dst[f];
        }
        scal[0] = ws; scal[1] = wd;
    }
}

// ---------------------------------------------------------------------------
// K1: per-edge segment max of leaky_relu logits
// ---------------------------------------------------------------------------
__global__ __launch_bounds__(256) void edge_max_kernel(
    const long long* __restrict__ ei, const float* __restrict__ xt,
    const float* __restrict__ scal, float* __restrict__ m)
{
    int idx = blockIdx.x * blockDim.x + threadIdx.x;
    if (idx >= ET * BS) return;
    int e  = idx >> 10;
    int bs = idx & (BS - 1);
    int src, dst;
    if (e < EE) { src = (int)ei[e]; dst = (int)ei[EE + e]; }
    else        { src = e - EE; dst = src; }
    float ws = scal[0], wd = scal[1];
    float xs = xt[src * BS + bs];
    float xd = xt[dst * BS + bs];
    float z  = xs * ws + xd * wd;
    float l  = (z > 0.0f) ? z : 0.2f * z;
    atomicMaxF(&m[dst * BS + bs], l);
}

// ---------------------------------------------------------------------------
// K2: accumulate den = sum e^{l-m}, num = sum e^{l-m} * x_src
// ---------------------------------------------------------------------------
__global__ __launch_bounds__(256) void edge_sum_kernel(
    const long long* __restrict__ ei, const float* __restrict__ xt,
    const float* __restrict__ scal, const float* __restrict__ m,
    float* __restrict__ den, float* __restrict__ num)
{
    int idx = blockIdx.x * blockDim.x + threadIdx.x;
    if (idx >= ET * BS) return;
    int e  = idx >> 10;
    int bs = idx & (BS - 1);
    int src, dst;
    if (e < EE) { src = (int)ei[e]; dst = (int)ei[EE + e]; }
    else        { src = e - EE; dst = src; }
    float ws = scal[0], wd = scal[1];
    float xs = xt[src * BS + bs];
    float xd = xt[dst * BS + bs];
    float z  = xs * ws + xd * wd;
    float l  = (z > 0.0f) ? z : 0.2f * z;
    int o = dst * BS + bs;
    float ex = expf(l - m[o]);
    atomicAdd(&den[o], ex);
    atomicAdd(&num[o], ex * xs);
}

// ---------------------------------------------------------------------------
// K3: build bf16 X  [BS][KDIM]:  X[bs, n*32+f] = relu(s*gw[f]+gb[f])
// ---------------------------------------------------------------------------
__global__ __launch_bounds__(256) void build_x_kernel(
    const float* __restrict__ num, const float* __restrict__ den,
    const float* __restrict__ gat_w, const float* __restrict__ gat_b,
    unsigned short* __restrict__ Xb)
{
    int idx = blockIdx.x * blockDim.x + threadIdx.x;
    if (idx >= BS * KDIM) return;
    int bs = idx / KDIM;
    int r  = idx - bs * KDIM;
    int n  = r >> 5;
    int f  = r & 31;
    float s   = num[n * BS + bs] / den[n * BS + bs];
    float val = fmaxf(s * gat_w[f] + gat_b[f], 0.0f);
    Xb[idx] = f2bf(val);
}

// ---------------------------------------------------------------------------
// K4: generic f32 -> bf16 conversion
// ---------------------------------------------------------------------------
__global__ __launch_bounds__(256) void conv_bf16_kernel(
    const float* __restrict__ in, unsigned short* __restrict__ out, int n)
{
    int idx = blockIdx.x * blockDim.x + threadIdx.x;
    if (idx >= n) return;
    out[idx] = f2bf(in[idx]);
}

// ---------------------------------------------------------------------------
// K5: LDS-staged bf16 WMMA GEMM  pre[z][BS][GDIM] = Xb[M,Kz] * Wb[N,Kz]^T
//     512 threads = 16 waves; macro tile 128x128; wave tile 32x32 (4 WMMA).
//     Double-buffered LDS staging: per K-step each thread does one
//     global_load_b128 for A and one for B; fragments come from LDS.
//     Rows padded to TPAD=40 bf16 so A-fragment ds_load_b128 is conflict-free.
// ---------------------------------------------------------------------------
__global__ __launch_bounds__(512) void gemm_bf16_kernel(
    const unsigned short* __restrict__ Xb,
    const unsigned short* __restrict__ Wb,
    float* __restrict__ pre)
{
    __shared__ __align__(16) unsigned short a_sh[2][128 * TPAD];  // 2 x 10 KB
    __shared__ __align__(16) unsigned short b_sh[2][128 * TPAD];  // 2 x 10 KB

    const int K  = KDIM;
    const int NK = KH / 32;           // 100 K-steps
    int t    = threadIdx.x;           // 0..511
    int m0   = blockIdx.x * 128;
    int n0   = blockIdx.y * 128;
    int kz   = blockIdx.z;
    const int kbeg = kz * KH;

    // --- staging assignment: thread -> (row, 8-element chunk) ---
    int srow = t >> 2;                // 0..127
    int schk = (t & 3) * 8;           // 0,8,16,24
    const unsigned short* ga = Xb + (size_t)(m0 + srow) * K + kbeg + schk;
    const unsigned short* gb = Wb + (size_t)(n0 + srow) * K + kbeg + schk;
    int soff = srow * TPAD + schk;

    // --- compute assignment: wave -> 32x32 sub-tile ---
    int wave = t >> 5;                // 0..15
    int lane = t & 31;
    int lo   = lane & 15;
    int hi   = lane >> 4;
    int am   = (wave >> 2) * 32;      // A row base within macro tile
    int bn   = (wave & 3) * 32;       // B col base within macro tile

    v8f acc00 = {}, acc01 = {}, acc10 = {}, acc11 = {};

    // prologue: stage K-step 0 into buffer 0
    {
        uint4 ra = *(const uint4*)(ga);
        uint4 rb = *(const uint4*)(gb);
        *(uint4*)(&a_sh[0][soff]) = ra;
        *(uint4*)(&b_sh[0][soff]) = rb;
    }
    __syncthreads();

    for (int ks = 0; ks < NK; ++ks) {
        int buf = ks & 1;
        bool more = (ks + 1) < NK;

        uint4 ra, rb;
        if (more) {
            ra = *(const uint4*)(ga + (ks + 1) * 32);
            rb = *(const uint4*)(gb + (ks + 1) * 32);
        }

        // fragments from LDS
        ABFrag fa0, fa1, fb0, fb1;
        {
            const unsigned short* p0 = &a_sh[buf][(am      + lo) * TPAD + hi * 8];
            const unsigned short* p1 = &a_sh[buf][(am + 16 + lo) * TPAD + hi * 8];
            fa0.u[0] = *(const uint4*)(p0);
            fa0.u[1] = *(const uint4*)(p0 + 16);
            fa1.u[0] = *(const uint4*)(p1);
            fa1.u[1] = *(const uint4*)(p1 + 16);
            const unsigned short* q0 = &b_sh[buf][(bn      + lo) * TPAD + hi * 8];
            const unsigned short* q1 = &b_sh[buf][(bn + 16 + lo) * TPAD + hi * 8];
            fb0.u[0] = *(const uint4*)(q0);
            fb0.u[1] = *(const uint4*)(q0 + 16);
            fb1.u[0] = *(const uint4*)(q1);
            fb1.u[1] = *(const uint4*)(q1 + 16);
        }

        acc00 = __builtin_amdgcn_wmma_f32_16x16x32_bf16(false, fa0.v, false, fb0.v, (short)0, acc00, false, false);
        acc01 = __builtin_amdgcn_wmma_f32_16x16x32_bf16(false, fa0.v, false, fb1.v, (short)0, acc01, false, false);
        acc10 = __builtin_amdgcn_wmma_f32_16x16x32_bf16(false, fa1.v, false, fb0.v, (short)0, acc10, false, false);
        acc11 = __builtin_amdgcn_wmma_f32_16x16x32_bf16(false, fa1.v, false, fb1.v, (short)0, acc11, false, false);

        if (more) {
            *(uint4*)(&a_sh[buf ^ 1][soff]) = ra;
            *(uint4*)(&b_sh[buf ^ 1][soff]) = rb;
        }
        __syncthreads();
    }

    // D layout: VGPR v -> row (base + hi*8 + v), col (base + lo)
    float* outz = pre + (size_t)kz * BS * GDIM;
    int r0 = m0 + am + hi * 8;
    int c0 = n0 + bn + lo;
#pragma unroll
    for (int v = 0; v < 8; ++v) {
        float* ra = outz + (size_t)(r0 + v) * GDIM + c0;
        ra[0]  = acc00[v];
        ra[16] = acc01[v];
        float* rb = outz + (size_t)(r0 + 16 + v) * GDIM + c0;
        rb[0]  = acc10[v];
        rb[16] = acc11[v];
    }
}

// ---------------------------------------------------------------------------
// K5b: reduce KSPLIT partial pregate chunks into one buffer
// ---------------------------------------------------------------------------
__global__ __launch_bounds__(256) void reduce_pre_kernel(
    const float* __restrict__ pre, float* __restrict__ preR)
{
    int idx = blockIdx.x * blockDim.x + threadIdx.x;
    if (idx >= BS * GDIM) return;
    float s = 0.0f;
#pragma unroll
    for (int z = 0; z < KSPLIT; ++z)
        s += pre[(size_t)z * BS * GDIM + idx];
    preR[idx] = s;
}

// ---------------------------------------------------------------------------
// K6: WMMA LSTM scan. One block of 1024 threads = 32 waves.
//     Wave w owns gate columns [w*16, w*16+16). w_hh (bf16) lives in registers
//     as 4 pre-loaded B fragments for all 64 steps. h carried as bf16 in LDS
//     (padded stride), gates staged in LDS f32, cell state in registers.
//     preR rows indexed bs = b*SS + s.
// ---------------------------------------------------------------------------
__global__ __launch_bounds__(1024) void lstm_wmma_kernel(
    const float* __restrict__ preR,         // [BS][GDIM]
    const unsigned short* __restrict__ whh, // [GDIM][HIDDEN] bf16
    const float* __restrict__ b_ih, const float* __restrict__ b_hh,
    float* __restrict__ hlast)
{
    __shared__ __align__(16) unsigned short h_sh[BB * HSTR];   // bf16, padded rows
    __shared__ float gates_sh[BB * GDIM];                      // 32 KB

    int tid  = threadIdx.x;
    int wave = tid >> 5;          // 0..31 -> N tile
    int lane = tid & 31;
    int lo   = lane & 15;
    int hi   = lane >> 4;
    int g    = wave * 16 + lo;    // gate column owned by this lane

    // Preload B fragments: B[k, n] = whh[g, k], 4 K-tiles of 32
    ABFrag bf[4];
#pragma unroll
    for (int kt = 0; kt < 4; ++kt) {
        const unsigned short* bp = whh + (size_t)g * HIDDEN + kt * 32 + hi * 8;
        bf[kt].u[0] = *(const uint4*)(bp);
        bf[kt].u[1] = *(const uint4*)(bp + 16);
    }
    float bias = b_ih[g] + b_hh[g];

    // elementwise phase mapping: pairs p = tid, tid+1024 -> (b, j)
    int b0_ = tid >> 7,           j0 = tid & 127;
    int b1_ = (tid + 1024) >> 7,  j1 = tid & 127;
    float c0 = 0.0f, c1 = 0.0f;

    for (int i = tid; i < BB * HSTR; i += 1024) h_sh[i] = 0;
    __syncthreads();

    for (int s = 0; s < SS; ++s) {
        // --- phase 1: recurrent GEMM tile via WMMA ---
        v8f acc = {};
#pragma unroll
        for (int kt = 0; kt < 4; ++kt) {
            ABFrag a;   // A row = b = lo, K chunk per lane half
            const unsigned short* hp = h_sh + lo * HSTR + kt * 32 + hi * 8;
            a.u[0] = *(const uint4*)(hp);
            a.u[1] = *(const uint4*)(hp + 16);
            acc = __builtin_amdgcn_wmma_f32_16x16x32_bf16(false, a.v, false, bf[kt].v,
                                                          (short)0, acc, false, false);
        }
#pragma unroll
        for (int v = 0; v < 8; ++v) {
            int b = hi * 8 + v;
            gates_sh[b * GDIM + g] =
                acc[v] + bias + preR[(size_t)(b * SS + s) * GDIM + g];
        }
        __syncthreads();

        // --- phase 2: elementwise gate update (2 (b,j) pairs per thread) ---
        {
            const float* gr = gates_sh + b0_ * GDIM;
            float cn = sigmoidf_(gr[HIDDEN + j0]) * c0
                     + sigmoidf_(gr[j0]) * tanhf(gr[2 * HIDDEN + j0]);
            c0 = cn;
            float h = sigmoidf_(gr[3 * HIDDEN + j0]) * tanhf(cn);
            h_sh[b0_ * HSTR + j0] = f2bf(h);
            if (s == SS - 1) hlast[b0_ * HIDDEN + j0] = h;
        }
        {
            const float* gr = gates_sh + b1_ * GDIM;
            float cn = sigmoidf_(gr[HIDDEN + j1]) * c1
                     + sigmoidf_(gr[j1]) * tanhf(gr[2 * HIDDEN + j1]);
            c1 = cn;
            float h = sigmoidf_(gr[3 * HIDDEN + j1]) * tanhf(cn);
            h_sh[b1_ * HSTR + j1] = f2bf(h);
            if (s == SS - 1) hlast[b1_ * HIDDEN + j1] = h;
        }
        __syncthreads();
    }
}

// ---------------------------------------------------------------------------
// K7: head  out[b,n] = <h_last[b,:], head_w[n,:]> + head_b[n]
// ---------------------------------------------------------------------------
__global__ __launch_bounds__(256) void head_kernel(
    const float* __restrict__ hlast, const float* __restrict__ head_w,
    const float* __restrict__ head_b, float* __restrict__ out)
{
    int idx = blockIdx.x * blockDim.x + threadIdx.x;
    if (idx >= BB * NNODES) return;
    int b = idx / NNODES;
    int n = idx - b * NNODES;
    float acc = head_b[n];
    const float* hw = head_w + n * HIDDEN;
    const float* hl = hlast + b * HIDDEN;
    for (int k = 0; k < HIDDEN; ++k) acc = fmaf(hl[k], hw[k], acc);
    out[idx] = acc;
}

// ---------------------------------------------------------------------------
// Host-side launch
// ---------------------------------------------------------------------------
extern "C" void kernel_launch(void* const* d_in, const int* in_sizes, int n_in,
                              void* d_out, int out_size, void* d_ws, size_t ws_size,
                              hipStream_t stream) {
    const float*      x      = (const float*)d_in[0];
    const long long*  ei     = (const long long*)d_in[1];
    const float*      gat_w  = (const float*)d_in[2];
    const float*      a_src  = (const float*)d_in[3];
    const float*      a_dst  = (const float*)d_in[4];
    const float*      gat_b  = (const float*)d_in[5];
    const float*      w_ih   = (const float*)d_in[6];
    const float*      w_hh   = (const float*)d_in[7];
    const float*      b_ih   = (const float*)d_in[8];
    const float*      b_hh   = (const float*)d_in[9];
    const float*      head_w = (const float*)d_in[10];
    const float*      head_b = (const float*)d_in[11];
    float*            out    = (float*)d_out;

    // workspace partition (256B aligned)
    char* p = (char*)d_ws;
    auto take = [&](size_t bytes) -> char* {
        char* r = p;
        p += (bytes + 255) & ~(size_t)255;
        return r;
    };
    float*          xt    = (float*)take(sizeof(float) * NNODES * BS);
    float*          m     = (float*)take(sizeof(float) * NNODES * BS);
    float*          den   = (float*)take(sizeof(float) * NNODES * BS);
    float*          num   = (float*)take(sizeof(float) * NNODES * BS);
    float*          scal  = (float*)take(sizeof(float) * 2);
    unsigned short* Xb    = (unsigned short*)take(sizeof(unsigned short) * (size_t)BS * KDIM);
    unsigned short* Wb    = (unsigned short*)take(sizeof(unsigned short) * (size_t)GDIM * KDIM);
    unsigned short* Whhb  = (unsigned short*)take(sizeof(unsigned short) * GDIM * HIDDEN);
    float*          pre   = (float*)take(sizeof(float) * (size_t)KSPLIT * BS * GDIM);
    float*          preR  = (float*)take(sizeof(float) * (size_t)BS * GDIM);
    float*          hlast = (float*)take(sizeof(float) * BB * HIDDEN);

    init_kernel<<<(NNODES * BS + 255) / 256, 256, 0, stream>>>(
        x, gat_w, a_src, a_dst, xt, m, den, num, scal);

    edge_max_kernel<<<(ET * BS + 255) / 256, 256, 0, stream>>>(ei, xt, scal, m);

    edge_sum_kernel<<<(ET * BS + 255) / 256, 256, 0, stream>>>(ei, xt, scal, m, den, num);

    build_x_kernel<<<((size_t)BS * KDIM + 255) / 256, 256, 0, stream>>>(
        num, den, gat_w, gat_b, Xb);

    conv_bf16_kernel<<<(GDIM * KDIM + 255) / 256, 256, 0, stream>>>(w_ih, Wb, GDIM * KDIM);
    conv_bf16_kernel<<<(GDIM * HIDDEN + 255) / 256, 256, 0, stream>>>(w_hh, Whhb, GDIM * HIDDEN);

    dim3 ggrid(BS / 128, GDIM / 128, KSPLIT);   // (8, 4, 5)
    gemm_bf16_kernel<<<ggrid, 512, 0, stream>>>(Xb, Wb, pre);

    reduce_pre_kernel<<<(BS * GDIM + 255) / 256, 256, 0, stream>>>(pre, preR);

    lstm_wmma_kernel<<<1, 1024, 0, stream>>>(preR, Whhb, b_ih, b_hh, hlast);

    head_kernel<<<(BB * NNODES + 255) / 256, 256, 0, stream>>>(hlast, head_w, head_b, out);
}
